// SmallLatentReasoner_8117488189883
// MI455X (gfx1250) — matrix operand order repaired and checked by reading, not compiled, not run on hardware
//
#include <hip/hip_runtime.h>
#include <hip/hip_bf16.h>

#define H 1024
#define B 512
#define STEPS 64
#define LDSROW (H + 16)  // +16 bf16 pad: 2080B row stride, 32B-aligned rows

typedef __attribute__((ext_vector_type(16))) __bf16 v16bf;
typedef __attribute__((ext_vector_type(8)))  __bf16 v8bf;
typedef __attribute__((ext_vector_type(8)))  float  v8f;

static __device__ __forceinline__ v8f wmma_bf16(v16bf a, v16bf b, v8f c) {
  // 8 args: (neg_a, A, neg_b, B, c_mod, C, reuse_a, reuse_b)
  return __builtin_amdgcn_wmma_f32_16x16x32_bf16(false, a, false, b, (short)0, c,
                                                 false, false);
}

// Register-placement-only concat of two v8bf halves (no per-element VALU).
static __device__ __forceinline__ v16bf concat8(v8bf lo, v8bf hi) {
  return __builtin_shufflevector(lo, hi, 0, 1, 2, 3, 4, 5, 6, 7,
                                 8, 9, 10, 11, 12, 13, 14, 15);
}

// Hardware-rate activations: v_exp_f32 + v_rcp_f32 instead of libm sequences.
static __device__ __forceinline__ float fast_sigmoid(float x) {
  return __builtin_amdgcn_rcpf(1.0f + __expf(-x));
}
static __device__ __forceinline__ float fast_tanh(float x) {
  float e = __expf(-2.0f * fabsf(x));  // e in (0,1]: no overflow path
  float t = (1.0f - e) * __builtin_amdgcn_rcpf(1.0f + e);
  return copysignf(t, x);
}

// Convert weights / initial state to bf16; build summed r/z weights
// (w_ih + w_hh for gate rows 0..2H, valid because x==h for all steps>=1).
__global__ void prep_kernel(const float* __restrict__ init_state,
                            const float* __restrict__ w_ih,
                            const float* __restrict__ w_hh,
                            const float* __restrict__ w_out,
                            __bf16* __restrict__ whh_b,
                            __bf16* __restrict__ wsum_rz,
                            __bf16* __restrict__ wihn_b,
                            __bf16* __restrict__ wout_b,
                            __bf16* __restrict__ h0_b) {
  int idx = blockIdx.x * blockDim.x + threadIdx.x;
  if (idx < 3 * H * H) whh_b[idx] = (__bf16)w_hh[idx];
  if (idx < 2 * H * H) wsum_rz[idx] = (__bf16)(w_ih[idx] + w_hh[idx]);
  if (idx < H * H) {
    wihn_b[idx] = (__bf16)w_ih[2 * H * H + idx];
    wout_b[idx] = (__bf16)w_out[idx];
  }
  if (idx < B * H) h0_b[idx] = (__bf16)init_state[idx];
}

// One GRU step. Block = 8 waves sharing one 16-column tile jt of H; the four
// 16xH weight panels (r, z, n_h, n_i) for that jt are staged into LDS once and
// reused by all 8 waves (8x cut in L2 weight traffic). Each wave owns one
// 16x16 tile of [B x H]: K-loop of 32 does 2 global A loads + 4 aligned 32B
// LDS B loads + 4 (or 3) WMMAs, branch-free (HAS_NI is compile-time).
template <bool HAS_NI>
__global__ void gru_step_kernel(const __bf16* __restrict__ hA,
                                const float* __restrict__ hPrevF, int prevStride,
                                const __bf16* __restrict__ Brz,   // [2H x H]
                                const __bf16* __restrict__ Bni,   // [H x H]
                                const __bf16* __restrict__ Bnh,   // [H x H]
                                const float* __restrict__ b_ih,
                                const float* __restrict__ b_hh,
                                float* __restrict__ hOutF, int outStride,
                                __bf16* __restrict__ hOutB) {
  extern __shared__ __bf16 smem[];  // 4 panels x 16 x LDSROW bf16 (~130 KB)

  const int tid = threadIdx.x;
  const int lane = tid & 31;
  const int wl = tid >> 5;             // wave in block: 0..7 -> row sub-tile
  const int jt = blockIdx.x & 63;      // column tile of H (shared by block)
  const int mblk = blockIdx.x >> 6;    // 0..3
  const int rowbase = (mblk * 8 + wl) * 16;
  const int l16 = lane & 15;
  const int hihalf = lane >> 4;        // 0: K 0..15 ; 1: K 16..31
  const int ncol = jt * 16 + l16;
  const int aoff = hihalf * 16;
  const int koff = hihalf * 16;
  const int arow = rowbase + l16;

  // ---- Stage B panels: rows jt*16..jt*16+15 of each gate matrix -> LDS ----
  {
    const int chunksPerPanel = 16 * (H / 8);  // 2048 v8bf chunks per panel
#pragma unroll
    for (int g = 0; g < 4; ++g) {
      const __bf16* src = (g == 0) ? Brz
                        : (g == 1) ? (Brz + (size_t)H * H)
                        : (g == 2) ? Bnh : Bni;
      __bf16* dstPanel = smem + g * 16 * LDSROW;
      for (int c = tid; c < chunksPerPanel; c += 256) {
        int row = c >> 7;   // /(H/8)
        int col8 = c & 127; // %(H/8)
        v8bf v = *(const v8bf*)(src + (size_t)(jt * 16 + row) * H + col8 * 8);
        *(v8bf*)(dstPanel + row * LDSROW + col8 * 8) = v;
      }
    }
  }
  __syncthreads();

  v8f acc_r = {}, acc_z = {}, acc_ni = {}, acc_nh = {};

  const __bf16* aRow = hA + arow * H;
  const __bf16* pr = smem + 0 * 16 * LDSROW + l16 * LDSROW + koff;
  const __bf16* pz = smem + 1 * 16 * LDSROW + l16 * LDSROW + koff;
  const __bf16* ph = smem + 2 * 16 * LDSROW + l16 * LDSROW + koff;
  const __bf16* pi = smem + 3 * 16 * LDSROW + l16 * LDSROW + koff;

#pragma unroll 8
  for (int kb = 0; kb < H; kb += 32) {
    // A tile 16x32 bf16: lanes 0-15 K{0..15}, lanes 16-31 K{16..31}
    v8bf alo = *(const v8bf*)(aRow + kb + aoff);
    v8bf ahi = *(const v8bf*)(aRow + kb + 8 + aoff);
    v16bf a = concat8(alo, ahi);

    // B tiles from LDS: lane n=l16 reads 16 contiguous K, 32B-aligned
    v16bf br = *(const v16bf*)(pr + kb);
    v16bf bz = *(const v16bf*)(pz + kb);
    v16bf bh = *(const v16bf*)(ph + kb);
    acc_r  = wmma_bf16(a, br, acc_r);
    acc_z  = wmma_bf16(a, bz, acc_z);
    acc_nh = wmma_bf16(a, bh, acc_nh);
    if (HAS_NI) {
      v16bf bi = *(const v16bf*)(pi + kb);
      acc_ni = wmma_bf16(a, bi, acc_ni);
    }
  }

  const float bir = b_ih[ncol],          bhr = b_hh[ncol];
  const float biz = b_ih[H + ncol],      bhz = b_hh[H + ncol];
  const float bin_ = b_ih[2 * H + ncol], bhn = b_hh[2 * H + ncol];

  // D layout: vgpr r, lane -> (M = r + 8*hihalf, N = l16)
#pragma unroll
  for (int r = 0; r < 8; ++r) {
    const int row = rowbase + r + hihalf * 8;
    float rg = fast_sigmoid(acc_r[r] + bir + bhr);
    float zg = fast_sigmoid(acc_z[r] + biz + bhz);
    float ni = HAS_NI ? (acc_ni[r] + bin_) : bin_;
    float ng = fast_tanh(ni + rg * (acc_nh[r] + bhn));
    float hp = hPrevF[(size_t)row * prevStride + ncol];
    float hn = (1.0f - zg) * ng + zg * hp;
    hOutF[(size_t)row * outStride + ncol] = hn;
    hOutB[row * H + ncol] = (__bf16)hn;
  }
}

// out = h_last @ w_out.T + b_out (single call; global-streamed is fine)
__global__ void linear_kernel(const __bf16* __restrict__ hA,
                              const __bf16* __restrict__ Bw,
                              const float* __restrict__ b_out,
                              float* __restrict__ out) {
  const int lane = threadIdx.x & 31;
  const int wave = blockIdx.x * (blockDim.x >> 5) + (threadIdx.x >> 5);
  const int mt = wave >> 6;
  const int jt = wave & 63;
  const int l16 = lane & 15;
  const int hihalf = lane >> 4;
  const int rowbase = mt * 16;
  const int ncol = jt * 16 + l16;
  const int aoff = hihalf * 16;
  const int koff = hihalf * 16;
  const int arow = rowbase + l16;

  v8f acc = {};
  const __bf16* aRow = hA + arow * H;
  const __bf16* bRow = Bw + ncol * H + koff;

  for (int kb = 0; kb < H; kb += 32) {
    v8bf alo = *(const v8bf*)(aRow + kb + aoff);
    v8bf ahi = *(const v8bf*)(aRow + kb + 8 + aoff);
    v16bf a = concat8(alo, ahi);
    v8bf blo = *(const v8bf*)(bRow + kb);
    v8bf bhi = *(const v8bf*)(bRow + kb + 8);
    v16bf b = concat8(blo, bhi);
    acc = wmma_bf16(a, b, acc);
  }

  const float bo = b_out[ncol];
#pragma unroll
  for (int r = 0; r < 8; ++r) {
    const int row = rowbase + r + hihalf * 8;
    out[row * H + ncol] = acc[r] + bo;
  }
}

extern "C" void kernel_launch(void* const* d_in, const int* in_sizes, int n_in,
                              void* d_out, int out_size, void* d_ws, size_t ws_size,
                              hipStream_t stream) {
  (void)in_sizes; (void)n_in; (void)out_size; (void)ws_size;
  const float* init_state = (const float*)d_in[0];
  const float* w_ih = (const float*)d_in[1];
  const float* w_hh = (const float*)d_in[2];
  const float* b_ih = (const float*)d_in[3];
  const float* b_hh = (const float*)d_in[4];
  const float* w_out = (const float*)d_in[5];
  const float* b_out = (const float*)d_in[6];
  // d_in[7] = steps (device scalar); the reference fixes STEPS=64.

  float* stacked = (float*)d_out;                   // [B, STEPS, H]
  float* outlin = stacked + (size_t)B * STEPS * H;  // [B, H]

  // Workspace: ~16.8 MB of bf16 staging
  char* w = (char*)d_ws;
  __bf16* whh_b = (__bf16*)w; w += (size_t)3 * H * H * sizeof(__bf16);
  __bf16* wsum  = (__bf16*)w; w += (size_t)2 * H * H * sizeof(__bf16);
  __bf16* wihn  = (__bf16*)w; w += (size_t)H * H * sizeof(__bf16);
  __bf16* woutb = (__bf16*)w; w += (size_t)H * H * sizeof(__bf16);
  __bf16* hb[2];
  hb[0] = (__bf16*)w; w += (size_t)B * H * sizeof(__bf16);
  hb[1] = (__bf16*)w; w += (size_t)B * H * sizeof(__bf16);

  const size_t ldsBytes = (size_t)4 * 16 * LDSROW * sizeof(__bf16);  // ~130 KB

  prep_kernel<<<(3 * H * H + 255) / 256, 256, 0, stream>>>(
      init_state, w_ih, w_hh, w_out, whh_b, wsum, wihn, woutb, hb[0]);

  // Step 0: x = 0  =>  gi = b_ih; gh from h0 @ w_hh.T (Brz = w_hh, no n_i GEMM)
  gru_step_kernel<false><<<256, 256, ldsBytes, stream>>>(
      hb[0], init_state, H,
      whh_b, whh_b + (size_t)2 * H * H, whh_b + (size_t)2 * H * H,
      b_ih, b_hh, stacked, STEPS * H, hb[1]);

  // Steps 1..63: x == h  =>  r/z use pre-summed weights (4 WMMAs / K-slice)
  for (int t = 1; t < STEPS; ++t) {
    gru_step_kernel<true><<<256, 256, ldsBytes, stream>>>(
        hb[t & 1], stacked + (size_t)(t - 1) * H, STEPS * H,
        wsum, wihn, whh_b + (size_t)2 * H * H,
        b_ih, b_hh, stacked + (size_t)t * H, STEPS * H, hb[(t + 1) & 1]);
  }

  // Final linear on h_last (bf16 copy lives in hb[STEPS & 1] == hb[0])
  linear_kernel<<<256, 256, 0, stream>>>(hb[STEPS & 1], woutb, b_out, outlin);
}
